// RGATLayer_88407606820908
// MI455X (gfx1250) — compile-verified, matrix operand order checked by reading, not run on hardware
//
#include <hip/hip_runtime.h>
#include <hip/hip_bf16.h>
#include <cstdint>
#include <cstddef>

typedef __attribute__((ext_vector_type(16))) __bf16 v16bf;
typedef __attribute__((ext_vector_type(8)))  float  v8f;

namespace {
constexpr int   NNODE = 40000;
constexpr int   NEDGE = 400000;
constexpr int   ENTOT = NEDGE + NNODE;   // edges + self loops
constexpr int   FIN   = 256;             // in feats (K)
constexpr int   NH    = 8;               // heads
constexpr int   NC    = 32;              // channels per head
constexpr int   HCH   = 256;             // out feats (N)
constexpr float SLOPE = 0.2f;

constexpr int BM = 128, BN = 128, BK = 32;
}

// ---------- helpers: monotonic float<->uint key for atomic max ----------
__device__ __forceinline__ unsigned fkey(float f) {
  unsigned b = __float_as_uint(f);
  return (b & 0x80000000u) ? ~b : (b | 0x80000000u);
}
__device__ __forceinline__ float funkey(unsigned k) {
  unsigned b = (k & 0x80000000u) ? (k & 0x7fffffffu) : ~k;
  return __uint_as_float(b);
}

// ---------- conversions ----------
__global__ void k_f32_to_bf16(const float* __restrict__ src,
                              __bf16* __restrict__ dst, int n) {
  int i = blockIdx.x * blockDim.x + threadIdx.x;
  int idx = i * 4;
  if (idx + 3 < n) {
    float4 v = *(const float4*)(src + idx);
    dst[idx + 0] = (__bf16)v.x; dst[idx + 1] = (__bf16)v.y;
    dst[idx + 2] = (__bf16)v.z; dst[idx + 3] = (__bf16)v.w;
  } else {
    for (int j = idx; j < n; ++j) dst[j] = (__bf16)src[j];
  }
}

// W [R][K][N] f32 -> Wt [R][N][K] bf16 (k-contiguous for WMMA B fragments)
__global__ void k_cvtW(const float* __restrict__ W, __bf16* __restrict__ wt) {
  int i = blockIdx.x * blockDim.x + threadIdx.x;
  if (i >= 4 * FIN * HCH) return;
  int r = i / (FIN * HCH);
  int rem = i - r * (FIN * HCH);
  int k = rem / HCH;
  int nn = rem - k * HCH;
  wt[((size_t)r * HCH + nn) * FIN + k] = (__bf16)W[i];
}

__global__ void k_fill_u32(unsigned* __restrict__ p, unsigned v, int n) {
  int i = blockIdx.x * blockDim.x + threadIdx.x;
  if (i < n) p[i] = v;
}

// out [2][3N][HCH]: paper=0.5*(bias1+bias3), author=bias0, subject=bias2
__global__ void k_init_out(float* __restrict__ out, const float* __restrict__ bias) {
  int i = blockIdx.x * blockDim.x + threadIdx.x;
  if (i >= 2 * 3 * NNODE * HCH) return;
  int c = i % HCH;
  int row = (i / HCH) % (3 * NNODE);
  int blk = row / NNODE;
  float v;
  if (blk == 0)      v = 0.5f * (bias[1 * HCH + c] + bias[3 * HCH + c]);
  else if (blk == 1) v = bias[0 * HCH + c];
  else               v = bias[2 * HCH + c];
  out[i] = v;
}

// ---------- WMMA GEMM: H[p] = X[type(p,copy)] @ W[p>>1],  [40000x256]x[256x256] ----------
__global__ void __launch_bounds__(256)
k_gemm(const __bf16* __restrict__ xb, const __bf16* __restrict__ wt,
       float* __restrict__ hout, int copy) {
  __shared__ __attribute__((aligned(16))) __bf16 As[BM][BK];
  __shared__ __attribute__((aligned(16))) __bf16 Bs[BN][BK];

  const int p = blockIdx.z;
  const int typeIdx = (p == 1 || p == 2) ? 1 : ((p == 5 || p == 6) ? 2 : 0);
  const __bf16* A = xb + (size_t)(copy * 3 + typeIdx) * NNODE * FIN;  // [N][K]
  const __bf16* B = wt + (size_t)(p >> 1) * HCH * FIN;                // [N_out][K]
  float* Hd = hout + (size_t)p * NNODE * HCH;

  const int m0 = blockIdx.x * BM;
  const int n0 = blockIdx.y * BN;
  const int tid = threadIdx.x;
  const int lane = tid & 31;
  const int wave = tid >> 5;
  const int wm = (wave >> 1) * 32;   // 0,32,64,96
  const int wn = (wave & 1) * 64;    // 0,64
  const int lrow = lane & 15;
  const int khalf = (lane >> 4) * 8; // lanes 0-15 -> K base 0, lanes 16-31 -> K base 8

  v8f zero = {};
  v8f acc[2][4];
#pragma unroll
  for (int mi = 0; mi < 2; ++mi)
#pragma unroll
    for (int ni = 0; ni < 4; ++ni) acc[mi][ni] = zero;

  for (int k0 = 0; k0 < FIN; k0 += BK) {
    // stage global -> LDS (each thread: 2 chunks of 8 bf16 for A and for B)
#pragma unroll
    for (int i = 0; i < 2; ++i) {
      int chunk = tid * 2 + i;     // 512 chunks cover 128x32
      int r = chunk >> 2;
      int c = (chunk & 3) * 8;
      int gr = m0 + r;
      if (gr >= NNODE) gr = NNODE - 1;
      *(uint4*)(&As[r][c]) = *(const uint4*)(A + (size_t)gr * FIN + k0 + c);
      *(uint4*)(&Bs[r][c]) = *(const uint4*)(B + (size_t)(n0 + r) * FIN + k0 + c);
    }
    __syncthreads();

    // fragments per ISA 16-bit A/B layout: elems[0..7]=K khalf..khalf+7,
    // elems[8..15]=K khalf+16..khalf+23
    union Frag { v16bf v; uint4 u[2]; };
    Frag fa[2], fb[4];
#pragma unroll
    for (int mi = 0; mi < 2; ++mi) {
      fa[mi].u[0] = *(const uint4*)(&As[wm + mi * 16 + lrow][khalf]);
      fa[mi].u[1] = *(const uint4*)(&As[wm + mi * 16 + lrow][khalf + 16]);
    }
#pragma unroll
    for (int ni = 0; ni < 4; ++ni) {
      fb[ni].u[0] = *(const uint4*)(&Bs[wn + ni * 16 + lrow][khalf]);
      fb[ni].u[1] = *(const uint4*)(&Bs[wn + ni * 16 + lrow][khalf + 16]);
    }
    __syncthreads();

#pragma unroll
    for (int mi = 0; mi < 2; ++mi)
#pragma unroll
      for (int ni = 0; ni < 4; ++ni)
        acc[mi][ni] = __builtin_amdgcn_wmma_f32_16x16x32_bf16(
            false, fa[mi].v, false, fb[ni].v, (short)0, acc[mi][ni],
            false, false);
  }

  // store D: lane col = lane&15, VGPR i -> row i + 8*(lane>>4)
#pragma unroll
  for (int mi = 0; mi < 2; ++mi) {
    int rbase = m0 + wm + mi * 16 + (lane >> 4) * 8;
#pragma unroll
    for (int ni = 0; ni < 4; ++ni) {
      int col = n0 + wn + ni * 16 + (lane & 15);
#pragma unroll
      for (int i = 0; i < 8; ++i) {
        int row = rbase + i;
        if (row < NNODE) Hd[(size_t)row * HCH + col] = acc[mi][ni][i];
      }
    }
  }
}

// ---------- node-level attention dots: s[p][n][h] = sum_c h[p][n][h][c]*att[h][c] ----------
__global__ void k_att(const float* __restrict__ hbuf, const float* __restrict__ att_src,
                      const float* __restrict__ att_dst, float* __restrict__ sbuf) {
  int i = blockIdx.x * blockDim.x + threadIdx.x;
  if (i >= 8 * NNODE * NH) return;
  int p = i / (NNODE * NH);
  int rem = i - p * (NNODE * NH);
  int n = rem / NH;
  int hh = rem - n * NH;
  int r = p >> 1;
  const float* a = ((p & 1) ? att_dst : att_src) + (r * NH + hh) * NC;
  const float* hp = hbuf + ((size_t)p * NNODE + n) * HCH + hh * NC;
  float acc = 0.f;
#pragma unroll
  for (int c = 0; c < NC; c += 4) {
    float4 hv = *(const float4*)(hp + c);
    float4 av = *(const float4*)(a + c);
    acc += hv.x * av.x + hv.y * av.y + hv.z * av.z + hv.w * av.w;
  }
  sbuf[i] = acc;
}

// ---------- edge pass 1: segment max via atomic max on ordered keys ----------
__global__ void k_emax(const float* __restrict__ ssrc, const float* __restrict__ sdst,
                       const int* __restrict__ ei, unsigned* __restrict__ mkey, int total) {
  int i = blockIdx.x * blockDim.x + threadIdx.x;
  if (i >= total) return;
  int e = i / NH, hh = i - e * NH;
  int s, d;
  if (e < NEDGE) { s = ei[e]; d = ei[NEDGE + e]; } else { s = d = e - NEDGE; }
  float v = ssrc[s * NH + hh] + sdst[d * NH + hh];
  v = v > 0.f ? v : v * SLOPE;
  atomicMax(mkey + d * NH + hh, fkey(v));
}

// ---------- edge pass 2: ex = exp(e - max), segment sum ----------
__global__ void k_esum(const float* __restrict__ ssrc, const float* __restrict__ sdst,
                       const int* __restrict__ ei, const unsigned* __restrict__ mkey,
                       float* __restrict__ exb, float* __restrict__ ssum, int total) {
  int i = blockIdx.x * blockDim.x + threadIdx.x;
  if (i >= total) return;
  int e = i / NH, hh = i - e * NH;
  int s, d;
  if (e < NEDGE) { s = ei[e]; d = ei[NEDGE + e]; } else { s = d = e - NEDGE; }
  float v = ssrc[s * NH + hh] + sdst[d * NH + hh];
  v = v > 0.f ? v : v * SLOPE;
  float m = funkey(mkey[d * NH + hh]);
  float t = __expf(v - m);
  exb[i] = t;
  atomicAdd(ssum + d * NH + hh, t);
}

// ---------- edge pass 3: out[dst] += scale * alpha * h_src[src], 64 lanes/edge ----------
__global__ void k_aggr(const float* __restrict__ hsrc, const int* __restrict__ ei,
                       const float* __restrict__ exb, const float* __restrict__ ssum,
                       float* __restrict__ oblk, float scale) {
  int i = blockIdx.x * blockDim.x + threadIdx.x;
  if (i >= ENTOT * 64) return;
  int t = i & 63;           // channels 4t..4t+3
  int e = i >> 6;
  int s, d;
  if (e < NEDGE) { s = ei[e]; d = ei[NEDGE + e]; } else { s = d = e - NEDGE; }
  int c0 = t * 4;
  int hh = c0 >> 5;
  float alpha = exb[e * NH + hh] / ssum[d * NH + hh] * scale;
  float4 hv = *(const float4*)(hsrc + (size_t)s * HCH + c0);
  float* o = oblk + (size_t)d * HCH + c0;
  atomicAdd(o + 0, alpha * hv.x);
  atomicAdd(o + 1, alpha * hv.y);
  atomicAdd(o + 2, alpha * hv.z);
  atomicAdd(o + 3, alpha * hv.w);
}

// ---------------------------------------------------------------------------
extern "C" void kernel_launch(void* const* d_in, const int* in_sizes, int n_in,
                              void* d_out, int out_size, void* d_ws, size_t ws_size,
                              hipStream_t stream) {
  (void)in_sizes; (void)n_in; (void)out_size; (void)ws_size;
  const float* x[6];
  for (int i = 0; i < 6; ++i) x[i] = (const float*)d_in[i];
  const float* W       = (const float*)d_in[6];
  const float* att_src = (const float*)d_in[7];
  const float* att_dst = (const float*)d_in[8];
  const float* bias    = (const float*)d_in[9];
  const int* ei[4];
  for (int i = 0; i < 4; ++i) ei[i] = (const int*)d_in[10 + i];
  float* out = (float*)d_out;

  // workspace carve (256B aligned)
  char* ws = (char*)d_ws;
  size_t off = 0;
  auto carve = [&](size_t bytes) -> char* {
    char* p = ws + off;
    off += (bytes + 255) & ~(size_t)255;
    return p;
  };
  __bf16*   xb   = (__bf16*)  carve((size_t)6 * NNODE * FIN * 2);   // bf16 features
  __bf16*   wt   = (__bf16*)  carve((size_t)4 * HCH * FIN * 2);     // bf16 W^T
  float*    hbuf = (float*)   carve((size_t)8 * NNODE * HCH * 4);   // 8 projections
  float*    sbuf = (float*)   carve((size_t)8 * NNODE * NH * 4);    // att dots
  unsigned* mkey = (unsigned*)carve((size_t)NNODE * NH * 4);        // segment max keys
  float*    ssum = (float*)   carve((size_t)NNODE * NH * 4);        // segment sums
  float*    exb  = (float*)   carve((size_t)ENTOT * NH * 4);        // per-edge exp

  // one-time conversions + output bias init
  {
    int n = NNODE * FIN;
    int blks = (n / 4 + 255) / 256;
    for (int i = 0; i < 6; ++i)
      k_f32_to_bf16<<<blks, 256, 0, stream>>>(x[i], xb + (size_t)i * n, n);
  }
  { int n = 4 * FIN * HCH; k_cvtW<<<(n + 255) / 256, 256, 0, stream>>>(W, wt); }
  { int n = 2 * 3 * NNODE * HCH; k_init_out<<<(n + 255) / 256, 256, 0, stream>>>(out, bias); }

  const int   dstBlk[4] = {1, 0, 2, 0};          // pa->author, ap->paper, ps->subject, sp->paper
  const float scl[4]    = {1.f, 0.5f, 1.f, 0.5f};

  for (int copy = 0; copy < 2; ++copy) {
    // 8 projections in one batched WMMA GEMM launch
    dim3 grid((NNODE + BM - 1) / BM, HCH / BN, 8);
    k_gemm<<<grid, 256, 0, stream>>>(xb, wt, hbuf, copy);

    { int n = 8 * NNODE * NH;
      k_att<<<(n + 255) / 256, 256, 0, stream>>>(hbuf, att_src, att_dst, sbuf); }

    for (int r = 0; r < 4; ++r) {
      const float* ssrc = sbuf + (size_t)(2 * r)     * NNODE * NH;
      const float* sdst = sbuf + (size_t)(2 * r + 1) * NNODE * NH;
      const float* hs   = hbuf + (size_t)(2 * r)     * NNODE * HCH;
      float* oblk = out + ((size_t)copy * 3 + dstBlk[r]) * NNODE * HCH;

      { int n = NNODE * NH;
        k_fill_u32<<<(n + 255) / 256, 256, 0, stream>>>(mkey, 0u, n);
        k_fill_u32<<<(n + 255) / 256, 256, 0, stream>>>((unsigned*)ssum, 0u, n); }
      { int n = ENTOT * NH;
        k_emax<<<(n + 255) / 256, 256, 0, stream>>>(ssrc, sdst, ei[r], mkey, n);
        k_esum<<<(n + 255) / 256, 256, 0, stream>>>(ssrc, sdst, ei[r], mkey, exb, ssum, n); }
      { int n = ENTOT * 64;
        k_aggr<<<(n + 255) / 256, 256, 0, stream>>>(hs, ei[r], exb, ssum, oblk, scl[r]); }
    }
  }
}
